// QLayerQuantum_65481071403405
// MI455X (gfx1250) — matrix operand order repaired and checked by reading, not compiled, not run on hardware
//
#include <hip/hip_runtime.h>
#include <hip/hip_bf16.h>

typedef __attribute__((ext_vector_type(16))) _Float16 v16h;
typedef __attribute__((ext_vector_type(4)))  _Float16 v4h;
typedef __attribute__((ext_vector_type(8)))  float    v8f;
typedef __attribute__((ext_vector_type(4)))  float    v4f;

#define NQ        32
#define HID       1024
#define WAVES     4
#define THREADS   (WAVES * 32)
#define ROWS_WG   (WAVES * 16)   // 64 rows per workgroup
#define COLS_PH   512            // columns staged per phase
#define TILES_PH  32             // 16-wide column tiles per phase

// z = cumprod(cos(x), axis=-1) ; out = z @ W^T + b
// WMMA with A = W-tile (16 hidden x 32 K), B = z^T (32 K x 16 batch):
// D element (M=hidden_local, N=batch_local) -> each lane owns 8 CONSECUTIVE
// hidden columns of one batch row => 2x global_store_b128 per tile.
__global__ void __launch_bounds__(THREADS)
qlayer_wmma_kernel(const float* __restrict__ x,
                   const float* __restrict__ W,
                   const float* __restrict__ b,
                   float* __restrict__ out)
{
    __shared__ _Float16 sW[TILES_PH * 512];   // 32 KB : A-fragments of W (per col tile)
    __shared__ _Float16 sZ[WAVES * 512];      //  4 KB : B-fragments of z^T (per wave)
    __shared__ float    sBias[HID];           //  4 KB

    const int tid  = threadIdx.x;
    const int lane = tid & 31;
    const int wave = tid >> 5;
    const int hf   = lane >> 4;     // lane half (0/1)
    const int nl   = lane & 15;     // index within 16-wide tile dim

    // ---- bias -> LDS via async global->LDS copy (ASYNCcnt path) ----
    for (int i = tid; i < HID; i += THREADS) {
        unsigned ldsoff = (unsigned)(size_t)(sBias + i);
        const float* g  = b + i;
        asm volatile("global_load_async_to_lds_b32 %0, %1, off"
                     :: "v"(ldsoff), "v"(g) : "memory");
    }

    // ---- prefetch second-phase W region into caches ----
    {
        const char* pf = (const char*)(W + COLS_PH * NQ) + tid * 512;
        __builtin_prefetch(pf, 0, 1);
    }

    // ---- z = cumprod(cos(x)) for this wave's 16 rows; stage as B-fragments ----
    // B element (k, batch r): lane = ((k>>3)&1)*16 + r, f16 slot = (k&7) + ((k&16)>>1)
    const int rowBase = blockIdx.x * ROWS_WG + wave * 16;
    for (int r = 0; r < 16; ++r) {
        float c = cosf(x[(size_t)(rowBase + r) * NQ + lane]);  // lane k = cos(x[row][k])
        #pragma unroll
        for (int off = 1; off < 32; off <<= 1) {               // inclusive product scan
            float other = __shfl_up(c, off, 32);
            if (lane >= off) c *= other;
        }
        sZ[wave * 512 + (((lane >> 3) & 1) * 16 + r) * 16 + (lane & 7) + ((lane & 16) >> 1)] = (_Float16)c;
    }

    asm volatile("s_wait_asynccnt 0" ::: "memory");  // bias resident in LDS
    __syncthreads();

    // B fragment (z^T) : 32 contiguous bytes per lane, fixed for whole kernel
    const v16h zfrag = *(const v16h*)(sZ + wave * 512 + lane * 16);

    for (int phase = 0; phase < 2; ++phase) {
        if (phase) __syncthreads();   // all reads of previous sW done

        // ---- stage 512 columns of W (f32 -> f16) into A-fragment layout ----
        // A element (k, hidden n): lane = ((k>>3)&1)*16 + (n&15),
        //                          slot = (k&7) + ((k&16)>>1)
        // 4 consecutive k (4-aligned) => same lane, 4 consecutive slots.
        for (int idx = tid; idx < COLS_PH * 8; idx += THREADS) {
            const int nloc = idx >> 3;                 // 0..511
            const int k0   = (idx & 7) * 4;            // 0,4,...,28
            const v4f w4 = *(const v4f*)(W + (size_t)(phase * COLS_PH + nloc) * NQ + k0);
            const int t     = nloc >> 4;
            const int hfk   = (k0 >> 3) & 1;
            const int slot0 = (k0 & 7) + ((k0 & 16) >> 1);
            v4h h4;
            #pragma unroll
            for (int i = 0; i < 4; ++i) h4[i] = (_Float16)w4[i];
            *(v4h*)(sW + t * 512 + (hfk * 16 + (nloc & 15)) * 16 + slot0) = h4;
        }
        __syncthreads();

        // ---- 32 column tiles: one WMMA each; bias preloaded into C ----
        for (int t = 0; t < TILES_PH; ++t) {
            const v16h wfrag = *(const v16h*)(sW + t * 512 + lane * 16);
            const int  col0  = phase * COLS_PH + t * 16 + hf * 8;  // 8 consecutive cols

            // C init from bias (varies along M = hidden): contiguous 32B LDS read
            const v4f blo = *(const v4f*)(sBias + col0);
            const v4f bhi = *(const v4f*)(sBias + col0 + 4);
            v8f c;
            #pragma unroll
            for (int i = 0; i < 4; ++i) { c[i] = blo[i]; c[i + 4] = bhi[i]; }

            c = __builtin_amdgcn_wmma_f32_16x16x32_f16(
                    /*neg_a=*/false, wfrag,
                    /*neg_b=*/false, zfrag,
                    /*c_mod=*/(short)0, c,
                    /*reuse_a=*/false, /*reuse_b=*/false);

            // lane owns batch row (rowBase+nl), hidden cols col0..col0+7
            float* p = out + (size_t)(rowBase + nl) * HID + col0;
            v4f lo, hi;
            #pragma unroll
            for (int i = 0; i < 4; ++i) { lo[i] = c[i]; hi[i] = c[i + 4]; }
            *(v4f*)p       = lo;
            *(v4f*)(p + 4) = hi;
        }
    }
}

extern "C" void kernel_launch(void* const* d_in, const int* in_sizes, int n_in,
                              void* d_out, int out_size, void* d_ws, size_t ws_size,
                              hipStream_t stream) {
    const float* x = (const float*)d_in[0];   // [65536, 32]
    const float* W = (const float*)d_in[1];   // [1024, 32]
    const float* b = (const float*)d_in[2];   // [1024]
    float* out = (float*)d_out;               // [65536, 1024]

    const int B = in_sizes[0] / NQ;           // 65536
    dim3 grid(B / ROWS_WG), block(THREADS);
    qlayer_wmma_kernel<<<grid, block, 0, stream>>>(x, W, b, out);
}
